// LTXAttention_77859167142201
// MI455X (gfx1250) — compile-verified, matrix operand order checked by reading, not run on hardware
//
#include <hip/hip_runtime.h>

#define DIMC 2048
#define NHC  32
#define HDC  64
#define NC   2048
#define TDC  6144   // 3*DIM

typedef __attribute__((ext_vector_type(16))) __bf16 v16bf;
typedef __attribute__((ext_vector_type(8)))  float  v8f;
typedef unsigned int su4 __attribute__((ext_vector_type(4)));
typedef unsigned int su8 __attribute__((ext_vector_type(8)));

__device__ __forceinline__ unsigned short f2bf(float f) {
  unsigned u = __float_as_uint(f);
  return (unsigned short)((u + 0x7FFFu + ((u >> 16) & 1u)) >> 16);  // RNE
}

// K-pair placement for 16-bit WMMA A/B fragments (CDNA5 ISA 7.12.2):
// pair j (j=0..7) holds K = {2j (j<4) | 2j+8 (j>=4)} + 8*(lane>=16), K+1
__device__ __forceinline__ int kmap(int j, int half) {
  return ((j < 4) ? (2 * j) : (2 * j + 8)) + 8 * half;
}

// Build a 16(x)x32(k) bf16 fragment from an LDS tile laid out [row_or_col][k],
// k contiguous with stride `ld` elements. Works for A (row=M) and B (row=N).
__device__ __forceinline__ v16bf load_frag(const unsigned short* base, int row,
                                           int ld, int lane) {
  const int half = lane >> 4;
  const unsigned short* p = base + row * ld;
  union { unsigned int u[8]; v16bf v; } f;
#pragma unroll
  for (int j = 0; j < 8; ++j)
    f.u[j] = *(const unsigned int*)(p + kmap(j, half));
  return f.v;
}

__device__ __forceinline__ v8f wmma_bf16(v16bf a, v16bf b, v8f c) {
  return __builtin_amdgcn_wmma_f32_16x16x32_bf16(false, a, false, b, (short)0,
                                                 c, false, false);
}

__device__ __forceinline__ v8f zero8() {
  v8f z = {0.f, 0.f, 0.f, 0.f, 0.f, 0.f, 0.f, 0.f};
  return z;
}

// LDS byte offset of a generic pointer into __shared__ (flat aperture rule:
// the low 32 address bits).
__device__ __forceinline__ unsigned lds_off(const void* p) {
  return (unsigned)(unsigned long long)p;
}

// --------------------------------------------------------------------- TDM
// Issue a 2D tensor_load_to_lds (CDNA5 ISA ch.7/8): tile_d1 rows of tile_d0
// bf16 elements, global row stride `stride_d0` elements, optional LDS row
// padding (pad codes per D# group1). Scalar-issued DMA, tracked by TENSORcnt.
__device__ __forceinline__ void tdm_load_2d_bf16(
    unsigned lds_addr, const void* gaddr, unsigned tile_d0, unsigned tile_d1,
    unsigned long long stride_d0, unsigned pad_interval_code,
    unsigned pad_amount_code, unsigned pad_enable) {
  const unsigned long long ga = (unsigned long long)gaddr;
  su4 g0;
  g0[0] = 1u;                                                // count=1
  g0[1] = lds_addr;                                          // lds_addr
  g0[2] = (unsigned)ga;                                      // global_addr lo
  g0[3] = (unsigned)((ga >> 32) & 0x1FFFFFFu) | (2u << 30);  // hi25 | type=2
  su8 g1;
  g1[0] = (1u << 16) | (pad_enable << 20) | (pad_interval_code << 22) |
          (pad_amount_code << 25);                 // data_size=2B + padding
  const unsigned td0 = 0x7FFFFFFFu, td1 = 0x7FFFFFFFu;  // no OOB clipping
  g1[1] = (td0 & 0xFFFFu) << 16;                          // tensor_dim0 lo16
  g1[2] = ((td0 >> 16) & 0xFFFFu) | ((td1 & 0xFFFFu) << 16);
  g1[3] = ((td1 >> 16) & 0xFFFFu) | ((tile_d0 & 0xFFFFu) << 16);
  g1[4] = (tile_d1 & 0xFFFFu);                           // tile_dim1, dim2=0
  g1[5] = (unsigned)(stride_d0 & 0xFFFFFFFFull);         // dim0_stride lo32
  g1[6] = (unsigned)((stride_d0 >> 32) & 0xFFFFull);     // dim0_stride hi16
  g1[7] = 0u;
  asm volatile("tensor_load_to_lds %0, %1" ::"s"(g0), "s"(g1) : "memory");
}

// ---------------------------------------------------------------- conversion
__global__ __launch_bounds__(256) void cvt_bf16_kernel(
    const float* __restrict__ in, unsigned short* __restrict__ out, int n) {
  int i = 4 * (blockIdx.x * 256 + threadIdx.x);
  if (i >= n) return;
  float4 v = *(const float4*)(in + i);
  ushort4 o;
  o.x = f2bf(v.x); o.y = f2bf(v.y); o.z = f2bf(v.z); o.w = f2bf(v.w);
  *(ushort4*)(out + i) = o;
}

// ------------------------------------------------------------- per-head gate
__global__ __launch_bounds__(256) void gate_kernel(
    const float* __restrict__ x, const float* __restrict__ Wg,
    const float* __restrict__ bg, float* __restrict__ gate) {
  const int n = blockIdx.x * 8 + (threadIdx.x >> 5);
  const int h = threadIdx.x & 31;
  const float* xr = x + (size_t)n * DIMC;
  const float* wr = Wg + (size_t)h * DIMC;
  float s = 0.f;
#pragma unroll 4
  for (int d = 0; d < DIMC; ++d) s = fmaf(xr[d], wr[d], s);
  s += bg[h];
  gate[(size_t)n * NHC + h] = 2.f / (1.f + __expf(-s));
}

// ------------------------------------------------------------------ bf16 GEMM
// C[M,Nn] = A[M,K](bf16) x B[K,Nn](bf16) + bias; 128x128 tile, 8 waves,
// each wave 32x64 = 2x4 wmma accumulators. A tile staged by TDM (wave 0),
// B tile cooperatively transposed to [n][k]. mode1: residual + v*gate_mod.
__global__ __launch_bounds__(256) void gemm_bf16_kernel(
    const unsigned short* __restrict__ A, const unsigned short* __restrict__ B,
    const float* __restrict__ bias, float* __restrict__ C, int M, int Nn, int K,
    const float* __restrict__ residual, const float* __restrict__ gate_mod,
    int mode) {
  __shared__ unsigned short sA[128 * 40];  // [m][k], 16dw rows + 4dw pad
  __shared__ unsigned short sB[128 * 40];  // [n][k], transposed on store

  const int tid  = threadIdx.x;
  const int lane = tid & 31;
  const int l15  = lane & 15;
  const int half = lane >> 4;
  const int wave = tid >> 5;
  const int wm   = wave >> 1;   // 0..3: 32-row strip
  const int wn   = wave & 1;    // 0..1: 64-col strip
  const long bm  = (long)blockIdx.y * 128;
  const long bn  = (long)blockIdx.x * 128;

  v8f acc[2][4];
#pragma unroll
  for (int i = 0; i < 2; ++i)
#pragma unroll
    for (int j = 0; j < 4; ++j) acc[i][j] = zero8();

  const int bkr = tid >> 3, bcs = (tid & 7) * 16;  // B: 8 threads/row
  const unsigned sA_off = lds_off(sA);

  for (int kt = 0; kt < K; kt += 32) {
    // stage A 128x32 via Tensor Data Mover (one DMA per workgroup tile)
    if (wave == 0) {
      tdm_load_2d_bf16(sA_off, A + (size_t)bm * K + kt, /*tile_d0=*/32,
                       /*tile_d1=*/128, /*stride_d0=*/(unsigned long long)K,
                       /*pad_interval(16dw)=*/3, /*pad_amount(4dw)=*/3, 1);
    }
    {  // stage B 32x128, transposed to [n][k]
      const unsigned short* g = B + (size_t)(kt + bkr) * Nn + bn + bcs;
      uint4 b0 = *(const uint4*)g;
      uint4 b1 = *(const uint4*)(g + 8);
      const unsigned short* p0 = (const unsigned short*)&b0;
      const unsigned short* p1 = (const unsigned short*)&b1;
#pragma unroll
      for (int i = 0; i < 8; ++i) sB[(bcs + i) * 40 + bkr] = p0[i];
#pragma unroll
      for (int i = 0; i < 8; ++i) sB[(bcs + 8 + i) * 40 + bkr] = p1[i];
      if (kt + 32 < K)
        __builtin_prefetch(B + (size_t)(kt + 32 + bkr) * Nn + bn + bcs, 0, 0);
    }
    if (wave == 0) __builtin_amdgcn_s_wait_tensorcnt(0);
    __syncthreads();

    v16bf af[2], bf[4];
#pragma unroll
    for (int i = 0; i < 2; ++i)
      af[i] = load_frag(sA, wm * 32 + i * 16 + l15, 40, lane);
#pragma unroll
    for (int j = 0; j < 4; ++j)
      bf[j] = load_frag(sB, wn * 64 + j * 16 + l15, 40, lane);

#pragma unroll
    for (int i = 0; i < 2; ++i)
#pragma unroll
      for (int j = 0; j < 4; ++j)
        acc[i][j] = wmma_bf16(af[i], bf[j], acc[i][j]);

    __syncthreads();
  }

  // epilogue: C/D layout is VGPR r <-> row r + 8*half, lane&15 <-> column
#pragma unroll
  for (int i = 0; i < 2; ++i) {
#pragma unroll
    for (int j = 0; j < 4; ++j) {
      const long col = bn + wn * 64 + j * 16 + l15;
      const float bv = bias[col];
#pragma unroll
      for (int r = 0; r < 8; ++r) {
        const long row = bm + wm * 32 + i * 16 + r + 8 * half;
        const size_t idx = (size_t)row * Nn + col;
        const float v = acc[i][j][r] + bv;
        if (mode == 0) C[idx] = v;
        else           C[idx] = residual[idx] + v * gate_mod[col];
      }
    }
  }
}

// ------------------------------------------- RMSNorm + RoPE + head reshape
__global__ __launch_bounds__(256) void normrope_kernel(
    const float* __restrict__ qkv, const float* __restrict__ nqw,
    const float* __restrict__ nkw, const float* __restrict__ rc,
    const float* __restrict__ rs, unsigned short* __restrict__ qh,
    unsigned short* __restrict__ kh, unsigned short* __restrict__ vh) {
  const int n = blockIdx.x;
  const float* q = qkv + (size_t)n * TDC;
  const float* k = q + DIMC;
  const float* v = q + 2 * DIMC;
  __shared__ float red[2][8];
  float sq = 0.f, sk2 = 0.f;
  for (int e = threadIdx.x; e < DIMC; e += 256) {
    float a = q[e]; sq += a * a;
    float b = k[e]; sk2 += b * b;
  }
#pragma unroll
  for (int m = 16; m >= 1; m >>= 1) {
    sq  += __shfl_xor(sq, m, 32);
    sk2 += __shfl_xor(sk2, m, 32);
  }
  if ((threadIdx.x & 31) == 0) {
    red[0][threadIdx.x >> 5] = sq;
    red[1][threadIdx.x >> 5] = sk2;
  }
  __syncthreads();
  float tq = 0.f, tk = 0.f;
#pragma unroll
  for (int i = 0; i < 8; ++i) { tq += red[0][i]; tk += red[1][i]; }
  const float rq = rsqrtf(tq / (float)DIMC + 1e-6f);
  const float rk = rsqrtf(tk / (float)DIMC + 1e-6f);

  for (int e = threadIdx.x; e < DIMC; e += 256) {
    const int hh = e >> 6, d = e & 63;
    const float c = rc[n * HDC + d], s = rs[n * HDC + d];
    const int rot = (d < 32) ? (e + 32) : (e - 32);
    const float sgn = (d < 32) ? -1.f : 1.f;
    const float qn = q[e] * rq * nqw[e];
    const float qr = q[rot] * rq * nqw[rot];
    const float kn = k[e] * rk * nkw[e];
    const float kr = k[rot] * rk * nkw[rot];
    const size_t o = ((size_t)hh * NC + n) * HDC + d;
    qh[o] = f2bf(qn * c + sgn * qr * s);
    kh[o] = f2bf(kn * c + sgn * kr * s);
    vh[o] = f2bf(v[e]);
  }
}

// -------------------------------------------------------- flash attention
// grid(32 q-blocks, 32 heads), 128 threads (4 waves x 16 query rows).
__global__ __launch_bounds__(128) void attn_kernel(
    const unsigned short* __restrict__ qh, const unsigned short* __restrict__ kh,
    const unsigned short* __restrict__ vh, const float* __restrict__ gate,
    unsigned short* __restrict__ attnb) {
  __shared__ unsigned short sk[64 * 72];     // [key][d], 32dw rows + 4dw pad
  __shared__ unsigned short sv[64 * 72];     // [d][key] (transposed)
  __shared__ unsigned short sp[4][16 * 40];  // per-wave P tile [qrow][key]

  const int h    = blockIdx.y;
  const int tid  = threadIdx.x;
  const int lane = tid & 31;
  const int l15  = lane & 15;
  const int half = lane >> 4;
  const int wave = tid >> 5;
  const int qbase = blockIdx.x * 64 + wave * 16;
  const float scale = 0.125f;  // 1/sqrt(64)

  // Q fragments for this wave's 16 rows, held in registers all kernel
  v16bf qf[2];
  {
    const unsigned short* qrow = qh + ((size_t)h * NC + qbase + l15) * HDC;
#pragma unroll
    for (int c = 0; c < 2; ++c) {
      union { unsigned int u[8]; v16bf v; } f;
#pragma unroll
      for (int j = 0; j < 8; ++j)
        f.u[j] = *(const unsigned int*)(qrow + c * 32 + kmap(j, half));
      qf[c] = f.v;
    }
  }

  v8f acc[4];
#pragma unroll
  for (int nt = 0; nt < 4; ++nt) acc[nt] = zero8();
  float mrow[8], lrow[8];
#pragma unroll
  for (int r = 0; r < 8; ++r) { mrow[r] = -1e30f; lrow[r] = 0.f; }

  unsigned short* spw = sp[wave];
  const unsigned sk_off = lds_off(sk);

  for (int kb = 0; kb < NC; kb += 64) {
    // stage K tile via TDM (one DMA per workgroup); V cooperatively transposed
    if (wave == 0) {
      tdm_load_2d_bf16(sk_off, kh + ((size_t)h * NC + kb) * HDC, /*tile_d0=*/64,
                       /*tile_d1=*/64, /*stride_d0=*/HDC,
                       /*pad_interval(32dw)=*/4, /*pad_amount(4dw)=*/3, 1);
    }
    {
      const int key = tid >> 1, cs = (tid & 1) * 32;
      const unsigned short* gv = vh + ((size_t)h * NC + kb + key) * HDC + cs;
      unsigned short tmp[32];
#pragma unroll
      for (int i = 0; i < 4; ++i)
        *(uint4*)(tmp + i * 8) = *(const uint4*)(gv + i * 8);
#pragma unroll
      for (int i = 0; i < 32; ++i) sv[(cs + i) * 72 + key] = tmp[i];
    }
    if (wave == 0) __builtin_amdgcn_s_wait_tensorcnt(0);
    __syncthreads();

#pragma unroll
    for (int sb = 0; sb < 2; ++sb) {  // two 32-key sub-blocks
      v8f s0 = zero8(), s1 = zero8();
#pragma unroll
      for (int c = 0; c < 2; ++c) {  // head-dim chunks of 32
        v16bf k0 = load_frag(sk + c * 32, sb * 32 + l15,      72, lane);
        v16bf k1 = load_frag(sk + c * 32, sb * 32 + 16 + l15, 72, lane);
        s0 = wmma_bf16(qf[c], k0, s0);
        s1 = wmma_bf16(qf[c], k1, s1);
      }
      // online softmax over the 32 keys; row stats via 16-lane xor reduce
#pragma unroll
      for (int r = 0; r < 8; ++r) {
        float v0 = s0[r] * scale, v1 = s1[r] * scale;
        float mx = fmaxf(v0, v1);
        mx = fmaxf(mx, __shfl_xor(mx, 1, 32));
        mx = fmaxf(mx, __shfl_xor(mx, 2, 32));
        mx = fmaxf(mx, __shfl_xor(mx, 4, 32));
        mx = fmaxf(mx, __shfl_xor(mx, 8, 32));
        const float mnew = fmaxf(mrow[r], mx);
        const float p0 = __expf(v0 - mnew);
        const float p1 = __expf(v1 - mnew);
        float rsum = p0 + p1;
        rsum += __shfl_xor(rsum, 1, 32);
        rsum += __shfl_xor(rsum, 2, 32);
        rsum += __shfl_xor(rsum, 4, 32);
        rsum += __shfl_xor(rsum, 8, 32);
        const float alpha = __expf(mrow[r] - mnew);
        lrow[r] = lrow[r] * alpha + rsum;
        mrow[r] = mnew;
#pragma unroll
        for (int nt = 0; nt < 4; ++nt) acc[nt][r] *= alpha;
        const int prow = (r + 8 * half) * 40;
        spw[prow + l15]      = f2bf(p0);
        spw[prow + 16 + l15] = f2bf(p1);
      }
      // O += P(16x32) x V(32x64); LDS ops are in-order within a wave
      v16bf pf = load_frag(spw, l15, 40, lane);
#pragma unroll
      for (int nt = 0; nt < 4; ++nt) {
        v16bf vf = load_frag(sv + sb * 32, nt * 16 + l15, 72, lane);
        acc[nt] = wmma_bf16(pf, vf, acc[nt]);
      }
    }
    __syncthreads();
  }

  // epilogue: normalize, apply per-head gate, write (N,DIM) bf16
#pragma unroll
  for (int r = 0; r < 8; ++r) {
    const int row = qbase + r + 8 * half;
    const float g = gate[(size_t)row * NHC + h];
    const float s = g / lrow[r];
#pragma unroll
    for (int nt = 0; nt < 4; ++nt) {
      const int d = nt * 16 + l15;
      attnb[(size_t)row * DIMC + h * HDC + d] = f2bf(acc[nt][r] * s);
    }
  }
}

// ------------------------------------------------------------------- launch
extern "C" void kernel_launch(void* const* d_in, const int* in_sizes, int n_in,
                              void* d_out, int out_size, void* d_ws,
                              size_t ws_size, hipStream_t stream) {
  (void)in_sizes; (void)n_in; (void)out_size; (void)ws_size;
  const float* x     = (const float*)d_in[0];
  const float* Wqkv  = (const float*)d_in[1];
  const float* bqkv  = (const float*)d_in[2];
  const float* nqw   = (const float*)d_in[3];
  const float* nkw   = (const float*)d_in[4];
  const float* Wg    = (const float*)d_in[5];
  const float* bg    = (const float*)d_in[6];
  const float* Wout  = (const float*)d_in[7];
  const float* bout  = (const float*)d_in[8];
  const float* rc    = (const float*)d_in[9];
  const float* rs    = (const float*)d_in[10];
  const float* resid = (const float*)d_in[11];
  const float* gmod  = (const float*)d_in[12];
  float* out = (float*)d_out;

  char* ws = (char*)d_ws;
  size_t off = 0;
  auto carve = [&](size_t bytes) {
    char* p = ws + off;
    off += (bytes + 255) & ~(size_t)255;
    return p;
  };
  unsigned short* xb    = (unsigned short*)carve((size_t)NC * DIMC * 2);
  unsigned short* wqb   = (unsigned short*)carve((size_t)DIMC * TDC * 2);
  unsigned short* wob   = (unsigned short*)carve((size_t)DIMC * DIMC * 2);
  float*          qkv   = (float*)carve((size_t)NC * TDC * 4);
  float*          gate  = (float*)carve((size_t)NC * NHC * 4);
  unsigned short* qh    = (unsigned short*)carve((size_t)NHC * NC * HDC * 2);
  unsigned short* kh    = (unsigned short*)carve((size_t)NHC * NC * HDC * 2);
  unsigned short* vh    = (unsigned short*)carve((size_t)NHC * NC * HDC * 2);
  unsigned short* attnb = (unsigned short*)carve((size_t)NC * DIMC * 2);

  cvt_bf16_kernel<<<(NC * DIMC / 4 + 255) / 256, 256, 0, stream>>>(x, xb, NC * DIMC);
  cvt_bf16_kernel<<<(DIMC * TDC / 4 + 255) / 256, 256, 0, stream>>>(Wqkv, wqb, DIMC * TDC);
  cvt_bf16_kernel<<<(DIMC * DIMC / 4 + 255) / 256, 256, 0, stream>>>(Wout, wob, DIMC * DIMC);
  gate_kernel<<<NC / 8, 256, 0, stream>>>(x, Wg, bg, gate);
  gemm_bf16_kernel<<<dim3(TDC / 128, NC / 128), 256, 0, stream>>>(
      xb, wqb, bqkv, qkv, NC, TDC, DIMC, nullptr, nullptr, 0);
  normrope_kernel<<<NC, 256, 0, stream>>>(qkv, nqw, nkw, rc, rs, qh, kh, vh);
  attn_kernel<<<dim3(NC / 64, NHC), 128, 0, stream>>>(qh, kh, vh, gate, attnb);
  gemm_bf16_kernel<<<dim3(DIMC / 128, NC / 128), 256, 0, stream>>>(
      attnb, wob, bout, out, NC, DIMC, DIMC, resid, gmod, 1);
}
